// NTMNet_62706522522200
// MI455X (gfx1250) — matrix operand order repaired and checked by reading, not compiled, not run on hardware
//
#include <hip/hip_runtime.h>
#include <hip/hip_bf16.h>
#include <math.h>

typedef __attribute__((ext_vector_type(16))) _Float16 v16h;
typedef __attribute__((ext_vector_type(2)))  _Float16 h2v;
typedef __attribute__((ext_vector_type(8)))  float    v8f;

#define EPSV 1e-12f
#define Bsz 64
#define Lsz 16384
#define Csz 64
#define Dsz 256
#define Hsz 100
#define Fsz 320
#define CHUNKS 32
#define ROWS_PER_BLOCK (Lsz / CHUNKS)   /* 512 */

/* workspace offsets, in floats (total ~2.25M floats = ~9.0 MB) */
#define WS_H    0                          /* h: 64*100                */
#define WS_KR   6400                       /* kvec_r: 64*64            */
#define WS_KW   10496                      /* kvec_w: 64*64            */
#define WS_E    14592                      /* e: 64*64                 */
#define WS_A    18688                      /* a: 64*64                 */
#define WS_SR   22784                      /* s_r: 64                  */
#define WS_SW   22848                      /* s_w: 64                  */
#define WS_DR   22912                      /* dots_r: 64*16384         */
#define WS_DW   (WS_DR + Bsz*Lsz)          /* dots_w: 64*16384         */
#define WS_SS   (WS_DW + Bsz*Lsz)          /* sumsq partials: 64*32    */
#define WS_ST   (WS_SS + Bsz*CHUNKS)       /* stats: 64*6              */
#define WS_RP   (WS_ST + Bsz*6)            /* read partials: 64*32*64  */

/* branch-free device math: single v_exp_f32, correct at both tails */
__device__ __forceinline__ float fast_tanh(float x) {
  /* tanh(x) = 1 - 2/(exp(2x)+1); exp->inf => 1, exp->0 => -1 */
  return 1.f - 2.f / (__expf(2.f * x) + 1.f);
}
__device__ __forceinline__ float fast_sigmoid(float x) {
  return 1.f / (1.f + __expf(-x));
}

/* ---------------- Kernel 1: MLP h = tanh(tanh(feat@W1+b1)@W2+b2), WMMA f16 ---------------- */
__global__ __launch_bounds__(256) void k_mlp(
    const float* __restrict__ inputs, const float* __restrict__ reading,
    const float* __restrict__ W1, const float* __restrict__ b1,
    const float* __restrict__ W2, const float* __restrict__ b2,
    float* __restrict__ h_out)
{
  __shared__ _Float16 featH[64 * Fsz];         /* feat as f16, row stride 320 (40 KB) */
  __shared__ _Float16 h1H[64 * 128];           /* h1, K-padded 100->128 with zeros (16 KB) */
  const int tid  = threadIdx.x;
  const int lane = tid & 31;
  const int wave = tid >> 5;
  const int mloc = lane & 15;
  const int hi   = lane >> 4;                  /* 0 or 1 */

  /* branch-free coalesced staging: feat = concat(inputs, reading) -> f16 LDS */
  for (int idx = tid; idx < 64*Dsz; idx += 256) {
    const int m = idx >> 8, k = idx & 255;
    featH[m*Fsz + k] = (_Float16)inputs[idx];
  }
  for (int idx = tid; idx < 64*Csz; idx += 256) {
    const int m = idx >> 6, c = idx & 63;
    featH[m*Fsz + Dsz + c] = (_Float16)reading[idx];
  }
  for (int i = tid; i < 64*128/2; i += 256) ((unsigned int*)h1H)[i] = 0u;
  __syncthreads();

  /* GEMM1: feat(64x320) @ W1(320x100); N padded to 112 (garbage cols discarded), 10 K-steps */
  for (int t = wave; t < 28; t += 8) {
    const int m0 = (t / 7) * 16;
    const int n0 = (t % 7) * 16;
    const int m  = m0 + mloc;
    const int n  = n0 + mloc;
    const int nc = (n < Hsz) ? n : (Hsz - 1);   /* clamp address; garbage col never stored */
    v8f acc = {};
    for (int kk = 0; kk < 10; ++kk) {
      const int k0 = kk * 32;
      v16h av, bv;
      #pragma unroll
      for (int v = 0; v < 8; ++v) {
        /* A (16x32 f16): lane = M row; VGPR v holds K pair (even k -> 4B aligned LDS read) */
        const int ka = k0 + ((v < 4) ? 0 : 16) + hi*8 + (v & 3)*2;
        const h2v ap = *(const h2v*)&featH[m*Fsz + ka];
        av[2*v]   = ap.x;
        av[2*v+1] = ap.y;
        /* B (32x16 f16): lane = N col; VGPR v holds K pair; unconditional coalesced loads */
        const int kb = k0 + ((v < 4) ? 0 : 16) + hi*8 + (v & 3)*2;
        bv[2*v]   = (_Float16)W1[(kb    )*Hsz + nc];
        bv[2*v+1] = (_Float16)W1[(kb + 1)*Hsz + nc];
      }
      acc = __builtin_amdgcn_wmma_f32_16x16x32_f16(false, av, false, bv, (short)0, acc, false, false);
    }
    if (n < Hsz) {
      const float bias = b1[n];
      #pragma unroll
      for (int r = 0; r < 8; ++r) {
        const int mr = m0 + hi*8 + r;            /* C/D layout: lanes 0-15 -> M=r, 16-31 -> M=8+r */
        h1H[mr*128 + n] = (_Float16)fast_tanh(acc[r] + bias);
      }
    }
  }
  __syncthreads();

  /* GEMM2: h1(64x128 zero-padded K) @ W2(100x100); clamped B addresses, A zero-pad kills k>=100 */
  for (int t = wave; t < 28; t += 8) {
    const int m0 = (t / 7) * 16;
    const int n0 = (t % 7) * 16;
    const int m  = m0 + mloc;
    const int n  = n0 + mloc;
    const int nc = (n < Hsz) ? n : (Hsz - 1);
    v8f acc = {};
    for (int kk = 0; kk < 4; ++kk) {
      const int k0 = kk * 32;
      v16h av, bv;
      #pragma unroll
      for (int v = 0; v < 8; ++v) {
        const int ka = k0 + ((v < 4) ? 0 : 16) + hi*8 + (v & 3)*2;
        const h2v ap = *(const h2v*)&h1H[m*128 + ka];
        av[2*v]   = ap.x;
        av[2*v+1] = ap.y;
        const int kb  = k0 + ((v < 4) ? 0 : 16) + hi*8 + (v & 3)*2;
        const int kc0 = (kb     < Hsz) ? kb     : (Hsz - 1);  /* A k-pad is 0, so 0*finite = 0 */
        const int kc1 = (kb + 1 < Hsz) ? kb + 1 : (Hsz - 1);
        bv[2*v]   = (_Float16)W2[kc0*Hsz + nc];
        bv[2*v+1] = (_Float16)W2[kc1*Hsz + nc];
      }
      acc = __builtin_amdgcn_wmma_f32_16x16x32_f16(false, av, false, bv, (short)0, acc, false, false);
    }
    if (n < Hsz) {
      const float bias = b2[n];
      #pragma unroll
      for (int r = 0; r < 8; ++r) {
        const int mr = m0 + hi*8 + r;
        h_out[mr*Hsz + n] = fast_tanh(acc[r] + bias);
      }
    }
  }
}

/* ---------------- Kernel 2: addressing heads (kvec_r/w normalized, s_r/w, e, a) ---------------- */
__global__ __launch_bounds__(64) void k_addr(
    const float* __restrict__ hbuf,
    const float* __restrict__ Wk_r, const float* __restrict__ bk_r,
    const float* __restrict__ Ws_r, const float* __restrict__ bs_r,
    const float* __restrict__ Wk_w, const float* __restrict__ bk_w,
    const float* __restrict__ Ws_w, const float* __restrict__ bs_w,
    const float* __restrict__ We,   const float* __restrict__ be,
    const float* __restrict__ Wa,   const float* __restrict__ ba,
    float* __restrict__ kr_out, float* __restrict__ kw_out,
    float* __restrict__ e_out,  float* __restrict__ a_out,
    float* __restrict__ sr_out, float* __restrict__ sw_out)
{
  __shared__ float sh_h[Hsz];
  __shared__ float red[64];
  const int b = blockIdx.x, c = threadIdx.x;
  for (int k = c; k < Hsz; k += 64) sh_h[k] = hbuf[b*Hsz + k];
  __syncthreads();

  float kr = bk_r[c], kw = bk_w[c], ev = be[c], av = ba[c];
  for (int k = 0; k < Hsz; ++k) {
    const float hv = sh_h[k];
    kr += hv * Wk_r[k*Csz + c];
    kw += hv * Wk_w[k*Csz + c];
    ev += hv * We  [k*Csz + c];
    av += hv * Wa  [k*Csz + c];
  }
  float psr = 0.f, psw = 0.f;
  for (int k = c; k < Hsz; k += 64) { psr += sh_h[k]*Ws_r[k]; psw += sh_h[k]*Ws_w[k]; }

  auto reduce = [&](float v) -> float {
    red[c] = v; __syncthreads();
    for (int s = 32; s > 0; s >>= 1) { if (c < s) red[c] += red[c + s]; __syncthreads(); }
    const float r = red[0]; __syncthreads();
    return r;
  };
  const float nr = reduce(kr*kr);
  const float nw = reduce(kw*kw);
  const float dr = reduce(psr);
  const float dw = reduce(psw);

  kr_out[b*Csz + c] = kr / fmaxf(sqrtf(nr), EPSV);
  kw_out[b*Csz + c] = kw / fmaxf(sqrtf(nw), EPSV);
  e_out [b*Csz + c] = fast_sigmoid(ev);
  a_out [b*Csz + c] = fast_sigmoid(av);
  if (c == 0) {
    sr_out[b] = fast_sigmoid(dr + bs_r[0]);
    sw_out[b] = fast_sigmoid(dw + bs_w[0]);
  }
}

/* ---------------- Kernel 3: streaming pass #1 — dots_r/w + sumsq partials ---------------- */
__global__ __launch_bounds__(256) void k_dots(
    const float* __restrict__ memory,
    const float* __restrict__ kr, const float* __restrict__ kw,
    float* __restrict__ dr, float* __restrict__ dw, float* __restrict__ sspart)
{
  __shared__ float red[256];
  const int tid   = threadIdx.x;
  const int b     = blockIdx.x / CHUNKS;
  const int chunk = blockIdx.x % CHUNKS;
  const int lane  = tid & 31;
  const int wave  = tid >> 5;
  const int c4    = lane & 15;     /* float4 slot within the 64-elem row */
  const int rp    = lane >> 4;     /* row within pair */
  const float4 kr4 = *(const float4*)&kr[b*Csz + c4*4];
  const float4 kw4 = *(const float4*)&kw[b*Csz + c4*4];
  const size_t mbase = (size_t)b * Lsz * Csz;
  const int l0 = chunk * ROWS_PER_BLOCK;

  float ss = 0.f;
  for (int step = 0; step < ROWS_PER_BLOCK/16; ++step) {
    const int l = l0 + step*16 + wave*2 + rp;
    const float* mp = &memory[mbase + (size_t)l*Csz + c4*4];
    if (step + 1 < ROWS_PER_BLOCK/16) __builtin_prefetch(mp + 16*Csz, 0, 0);
    const float4 m4 = *(const float4*)mp;
    float pr = m4.x*kr4.x + m4.y*kr4.y + m4.z*kr4.z + m4.w*kr4.w;
    float pw = m4.x*kw4.x + m4.y*kw4.y + m4.z*kw4.z + m4.w*kw4.w;
    ss += m4.x*m4.x + m4.y*m4.y + m4.z*m4.z + m4.w*m4.w;
    pr += __shfl_xor(pr, 1); pr += __shfl_xor(pr, 2); pr += __shfl_xor(pr, 4); pr += __shfl_xor(pr, 8);
    pw += __shfl_xor(pw, 1); pw += __shfl_xor(pw, 2); pw += __shfl_xor(pw, 4); pw += __shfl_xor(pw, 8);
    if (c4 == 0) { dr[b*Lsz + l] = pr; dw[b*Lsz + l] = pw; }
  }
  red[tid] = ss; __syncthreads();
  for (int s = 128; s > 0; s >>= 1) { if (tid < s) red[tid] += red[tid + s]; __syncthreads(); }
  if (tid == 0) sspart[blockIdx.x] = red[0];
}

/* ---------------- Kernel 4: per-batch softmax stats (scale, max, 1/sum) ---------------- */
__global__ __launch_bounds__(256) void k_stats(
    const float* __restrict__ dr, const float* __restrict__ dw,
    const float* __restrict__ sspart,
    const float* __restrict__ sr, const float* __restrict__ sw,
    float* __restrict__ st)
{
  __shared__ float red[256];
  __shared__ float sc[2];
  const int b = blockIdx.x, tid = threadIdx.x;
  if (tid == 0) {
    float ssum = 0.f;
    for (int i = 0; i < CHUNKS; ++i) ssum += sspart[b*CHUNKS + i];
    const float mn = fmaxf(sqrtf(ssum), EPSV);
    sc[0] = sr[b] / mn; sc[1] = sw[b] / mn;
  }
  __syncthreads();
  const float scr = sc[0], scw = sc[1];
  const float* drb = dr + (size_t)b * Lsz;
  const float* dwb = dw + (size_t)b * Lsz;

  float mr = -1e30f, mw = -1e30f;
  for (int i = tid; i < Lsz; i += 256) { mr = fmaxf(mr, scr*drb[i]); mw = fmaxf(mw, scw*dwb[i]); }
  red[tid] = mr; __syncthreads();
  for (int s = 128; s > 0; s >>= 1) { if (tid < s) red[tid] = fmaxf(red[tid], red[tid+s]); __syncthreads(); }
  const float maxr = red[0]; __syncthreads();
  red[tid] = mw; __syncthreads();
  for (int s = 128; s > 0; s >>= 1) { if (tid < s) red[tid] = fmaxf(red[tid], red[tid+s]); __syncthreads(); }
  const float maxw = red[0]; __syncthreads();

  /* __expf here must match k_final's __expf so weights normalize consistently */
  float er = 0.f, ew = 0.f;
  for (int i = tid; i < Lsz; i += 256) { er += __expf(scr*drb[i] - maxr); ew += __expf(scw*dwb[i] - maxw); }
  red[tid] = er; __syncthreads();
  for (int s = 128; s > 0; s >>= 1) { if (tid < s) red[tid] += red[tid+s]; __syncthreads(); }
  const float sumr = red[0]; __syncthreads();
  red[tid] = ew; __syncthreads();
  for (int s = 128; s > 0; s >>= 1) { if (tid < s) red[tid] += red[tid+s]; __syncthreads(); }
  const float sumw = red[0];

  if (tid == 0) {
    float* s6 = st + b*6;
    s6[0] = scr; s6[1] = maxr; s6[2] = 1.f/sumr;
    s6[3] = scw; s6[4] = maxw; s6[5] = 1.f/sumw;
  }
}

/* ---------------- Kernel 5: streaming pass #2 — new_mem write + read partials ---------------- */
__global__ __launch_bounds__(256) void k_final(
    const float* __restrict__ memory,
    const float* __restrict__ dr, const float* __restrict__ dw,
    const float* __restrict__ st,
    const float* __restrict__ evec, const float* __restrict__ avec,
    float* __restrict__ newmem, float* __restrict__ readpart)
{
  __shared__ float4 accbuf[256];
  const int tid   = threadIdx.x;
  const int b     = blockIdx.x / CHUNKS;
  const int chunk = blockIdx.x % CHUNKS;
  const int lane  = tid & 31;
  const int wave  = tid >> 5;
  const int c4    = lane & 15;
  const int rp    = lane >> 4;
  const float scr = st[b*6+0], maxr = st[b*6+1], isr = st[b*6+2];
  const float scw = st[b*6+3], maxw = st[b*6+4], isw = st[b*6+5];
  const float4 e4 = *(const float4*)&evec[b*Csz + c4*4];
  const float4 a4 = *(const float4*)&avec[b*Csz + c4*4];
  const size_t mbase = (size_t)b * Lsz * Csz;
  const int l0 = chunk * ROWS_PER_BLOCK;

  float4 racc = {0.f, 0.f, 0.f, 0.f};
  for (int step = 0; step < ROWS_PER_BLOCK/16; ++step) {
    const int l = l0 + step*16 + wave*2 + rp;
    const float* mp = &memory[mbase + (size_t)l*Csz + c4*4];
    if (step + 1 < ROWS_PER_BLOCK/16) __builtin_prefetch(mp + 16*Csz, 0, 0);
    const float4 m4 = *(const float4*)mp;
    const float wr = __expf(scr*dr[b*Lsz + l] - maxr) * isr;   /* arg <= 0, no overflow */
    const float ww = __expf(scw*dw[b*Lsz + l] - maxw) * isw;
    float4 o;
    o.x = m4.x*(1.f - ww*e4.x) + ww*a4.x;
    o.y = m4.y*(1.f - ww*e4.y) + ww*a4.y;
    o.z = m4.z*(1.f - ww*e4.z) + ww*a4.z;
    o.w = m4.w*(1.f - ww*e4.w) + ww*a4.w;
    *(float4*)&newmem[mbase + (size_t)l*Csz + c4*4] = o;
    racc.x += wr*m4.x; racc.y += wr*m4.y; racc.z += wr*m4.z; racc.w += wr*m4.w;
  }
  accbuf[tid] = racc;
  __syncthreads();
  if (tid < 64) {
    const int cc4 = tid >> 2, comp = tid & 3;
    float s = 0.f;
    for (int i = 0; i < 16; ++i) s += ((const float*)&accbuf[cc4 + 16*i])[comp];
    readpart[(size_t)blockIdx.x * Csz + tid] = s;
  }
}

/* ---------------- Kernel 6: fixed-order reduce of read partials ---------------- */
__global__ __launch_bounds__(64) void k_red(
    const float* __restrict__ readpart, float* __restrict__ out)
{
  const int b = blockIdx.x, c = threadIdx.x;
  float s = 0.f;
  for (int i = 0; i < CHUNKS; ++i) s += readpart[(size_t)(b*CHUNKS + i)*Csz + c];
  out[b*Csz + c] = s;
}

extern "C" void kernel_launch(void* const* d_in, const int* in_sizes, int n_in,
                              void* d_out, int out_size, void* d_ws, size_t ws_size,
                              hipStream_t stream)
{
  const float* inputs  = (const float*)d_in[0];
  const float* reading = (const float*)d_in[1];
  const float* memory  = (const float*)d_in[2];
  const float* W1   = (const float*)d_in[3];
  const float* b1   = (const float*)d_in[4];
  const float* W2   = (const float*)d_in[5];
  const float* b2   = (const float*)d_in[6];
  const float* Wk_r = (const float*)d_in[7];
  const float* bk_r = (const float*)d_in[8];
  const float* Ws_r = (const float*)d_in[9];
  const float* bs_r = (const float*)d_in[10];
  const float* Wk_w = (const float*)d_in[11];
  const float* bk_w = (const float*)d_in[12];
  const float* Ws_w = (const float*)d_in[13];
  const float* bs_w = (const float*)d_in[14];
  const float* We   = (const float*)d_in[15];
  const float* be   = (const float*)d_in[16];
  const float* Wa   = (const float*)d_in[17];
  const float* ba   = (const float*)d_in[18];

  float* ws       = (float*)d_ws;
  float* out_read = (float*)d_out;               /* (B, C)    = 4096 floats  */
  float* out_mem  = out_read + Bsz*Csz;          /* (B, L, C) = 67.1M floats */

  k_mlp  <<<1, 256, 0, stream>>>(inputs, reading, W1, b1, W2, b2, ws + WS_H);
  k_addr <<<Bsz, 64, 0, stream>>>(ws + WS_H, Wk_r, bk_r, Ws_r, bs_r, Wk_w, bk_w, Ws_w, bs_w,
                                  We, be, Wa, ba,
                                  ws + WS_KR, ws + WS_KW, ws + WS_E, ws + WS_A,
                                  ws + WS_SR, ws + WS_SW);
  k_dots <<<Bsz*CHUNKS, 256, 0, stream>>>(memory, ws + WS_KR, ws + WS_KW,
                                          ws + WS_DR, ws + WS_DW, ws + WS_SS);
  k_stats<<<Bsz, 256, 0, stream>>>(ws + WS_DR, ws + WS_DW, ws + WS_SS,
                                   ws + WS_SR, ws + WS_SW, ws + WS_ST);
  k_final<<<Bsz*CHUNKS, 256, 0, stream>>>(memory, ws + WS_DR, ws + WS_DW, ws + WS_ST,
                                          ws + WS_E, ws + WS_A, out_mem, ws + WS_RP);
  k_red  <<<Bsz, 64, 0, stream>>>(ws + WS_RP, out_read);
}
